// CausalSelfAttention_68874095559369
// MI455X (gfx1250) — compile-verified, matrix operand order checked by reading, not run on hardware
//
#include <hip/hip_runtime.h>
#include <hip/hip_bf16.h>
#include <math.h>
#include <stdint.h>

#define T_SEQ   4096
#define DM      1024
#define NH      16
#define HD      64
#define WINDOW  1024
#define SCALE_F 0.12f
#define NEG_BIG (-1.0e30f)

typedef __bf16 bf16;
typedef __attribute__((ext_vector_type(16))) __bf16 v16bf;
typedef __attribute__((ext_vector_type(8)))  __bf16 v8bf;
typedef __attribute__((ext_vector_type(8)))  float  v8f;

// ---------------------------------------------------------------------------
// Fragment loader for 16-bit WMMA operands (A 16x32 MxK row-major, or B 32x16
// KxN where the "row pointer" is the K-contiguous row of the N-major matrix).
// Lane half selects K sub-offset 0 or 8; VGPRs 0-3 hold K=base..base+7,
// VGPRs 4-7 hold K=16+base..16+base+7.  Two 16B loads.
// ---------------------------------------------------------------------------
__device__ __forceinline__ v16bf load_frag(const bf16* __restrict__ row, int k0, int lane) {
    const int base = (lane & 16) ? 8 : 0;
    const v8bf lo = *(const v8bf*)(row + k0 + base);
    const v8bf hi = *(const v8bf*)(row + k0 + 16 + base);
    v16bf r;
#pragma unroll
    for (int i = 0; i < 8; ++i) { r[i] = lo[i]; r[i + 8] = hi[i]; }
    return r;
}

__device__ __forceinline__ v8f wmma_bf16(v16bf a, v16bf b, v8f c) {
    return __builtin_amdgcn_wmma_f32_16x16x32_bf16(
        /*neg_a=*/false, a, /*neg_b=*/false, b,
        /*c_mod=*/(short)0, c, /*reuse_a=*/false, /*reuse_b=*/false);
}

// ---------------------------------------------------------------------------
// CDNA5 async global->LDS copy (ASYNCcnt-tracked), 8 bytes per lane so all
// 256 block threads participate (branch-free issue).  lds_off: per-lane LDS
// byte offset (low 32 bits of the flat shared address per the aperture rule).
// ---------------------------------------------------------------------------
__device__ __forceinline__ void async_ld_lds_b64(uint32_t lds_off, const void* gsrc) {
    asm volatile("global_load_async_to_lds_b64 %0, %1, off"
                 :: "v"(lds_off), "v"((uint64_t)(uintptr_t)gsrc)
                 : "memory");
}
template <int N>
__device__ __forceinline__ void wait_asynccnt() {
    asm volatile("s_wait_asynccnt %0" :: "n"(N) : "memory");
}

// ---------------------------------------------------------------------------
// fp32 -> bf16 convert
// ---------------------------------------------------------------------------
__global__ __launch_bounds__(256) void cvt_f32_bf16(const float* __restrict__ in,
                                                    bf16* __restrict__ out, int n) {
    int i = blockIdx.x * 256 + threadIdx.x;
    if (i < n) out[i] = (bf16)in[i];
}

// ---------------------------------------------------------------------------
// C[M][N] = A[M][K] x W[N][K]^T   (bf16 in, f32 out), K compile-time.
// Block: 256 threads = 8 waves, tile 32(M) x 256(N); wave tile 16 x 64.
// A tile (32x32 bf16, 2KB/step) staged in LDS via a peeled, double-buffered
// async pipeline (global_load_async_to_lds_b64 + s_wait_asynccnt); B streamed
// from L2 with global_load_b128 feeding WMMA.
// ---------------------------------------------------------------------------
template <int K>
__global__ __launch_bounds__(256) void gemm_bf16_wmma(
    const bf16* __restrict__ A, const bf16* __restrict__ W,
    float* __restrict__ C, int M, int N) {
    __shared__ __bf16 sA[2][32][32];
    const int tid  = threadIdx.x;
    const int lane = tid & 31;
    const int wave = tid >> 5;
    const int mblk = blockIdx.y * 32;
    const int m0 = mblk + (wave >> 2) * 16;
    const int n0 = blockIdx.x * 256 + (wave & 3) * 64;
    const int ln = lane & 15;
    const int base = (lane & 16) ? 8 : 0;

    v8f acc0 = {}, acc1 = {}, acc2 = {}, acc3 = {};
    const bf16* w0 = W + (size_t)(n0 +  0 + ln) * K;
    const bf16* w1 = W + (size_t)(n0 + 16 + ln) * K;
    const bf16* w2 = W + (size_t)(n0 + 32 + ln) * K;
    const bf16* w3 = W + (size_t)(n0 + 48 + ln) * K;

    // Async A-tile staging: thread t copies 8B; row = t>>3, 8B-seg = t&7.
    const int arow = tid >> 3;
    const int acol = (tid & 7) * 4;                 // bf16 elements
    const bf16* agsrc = A + (size_t)(mblk + arow) * K + acol;
    const __bf16* ar = &sA[0][(wave >> 2) * 16 + ln][0];   // buf stride 2048B

    constexpr int NIT = K / 32;

    async_ld_lds_b64((uint32_t)(uintptr_t)&sA[0][arow][acol], agsrc);

    for (int it = 0; it < NIT - 1; ++it) {
        const int k0 = it * 32;
        async_ld_lds_b64((uint32_t)(uintptr_t)&sA[(it + 1) & 1][arow][acol],
                         agsrc + (k0 + 32));
        wait_asynccnt<1>();
        __syncthreads();
        const __bf16* arc = ar + (size_t)(it & 1) * (32 * 32);
        v16bf a;
#pragma unroll
        for (int i = 0; i < 8; ++i) { a[i] = arc[base + i]; a[i + 8] = arc[16 + base + i]; }
        acc0 = wmma_bf16(a, load_frag(w0, k0, lane), acc0);
        acc1 = wmma_bf16(a, load_frag(w1, k0, lane), acc1);
        acc2 = wmma_bf16(a, load_frag(w2, k0, lane), acc2);
        acc3 = wmma_bf16(a, load_frag(w3, k0, lane), acc3);
        __syncthreads();
    }
    // Epilogue iteration.
    {
        const int k0 = (NIT - 1) * 32;
        wait_asynccnt<0>();
        __syncthreads();
        const __bf16* arc = ar + (size_t)((NIT - 1) & 1) * (32 * 32);
        v16bf a;
#pragma unroll
        for (int i = 0; i < 8; ++i) { a[i] = arc[base + i]; a[i + 8] = arc[16 + base + i]; }
        acc0 = wmma_bf16(a, load_frag(w0, k0, lane), acc0);
        acc1 = wmma_bf16(a, load_frag(w1, k0, lane), acc1);
        acc2 = wmma_bf16(a, load_frag(w2, k0, lane), acc2);
        acc3 = wmma_bf16(a, load_frag(w3, k0, lane), acc3);
    }
    const int rbase = m0 + ((lane & 16) ? 8 : 0);
#pragma unroll
    for (int vi = 0; vi < 8; ++vi) {
        float* crow = C + (size_t)(rbase + vi) * N + n0 + ln;
        crow[0] = acc0[vi]; crow[16] = acc1[vi]; crow[32] = acc2[vi]; crow[48] = acc3[vi];
    }
}

// ---------------------------------------------------------------------------
// RMSNorm(q,k) + RoPE(first 32 dims) + v*0.5, repack to head-major bf16.
// One wave per (token t, head h); lane owns dims d0=lane, d1=lane+32.
// qb,kb: [H][T][64];  vt: [H][64][T] (transposed for P*V B operand)
// ---------------------------------------------------------------------------
__global__ __launch_bounds__(256) void norm_rope_pack(
    const float* __restrict__ qkv, const float* __restrict__ qw,
    const float* __restrict__ kw, const float* __restrict__ cosT,
    const float* __restrict__ sinT, bf16* __restrict__ qb,
    bf16* __restrict__ kbuf, bf16* __restrict__ vt) {
    const int lane = threadIdx.x & 31;
    const int wave = threadIdx.x >> 5;
    const int p = blockIdx.x * 8 + wave;      // 0 .. T*H-1
    const int t = p >> 4;
    const int h = p & 15;
    const int d0 = lane, d1 = lane + 32;

    const float* row = qkv + (size_t)t * (3 * DM);
    float q0v = row[h * HD + d0],           q1v = row[h * HD + d1];
    float k0v = row[DM + h * HD + d0],      k1v = row[DM + h * HD + d1];
    float v0v = row[2*DM + h * HD + d0] * 0.5f;
    float v1v = row[2*DM + h * HD + d1] * 0.5f;

    float qs = q0v * q0v + q1v * q1v;
    float ks = k0v * k0v + k1v * k1v;
#pragma unroll
    for (int m = 16; m >= 1; m >>= 1) {
        qs += __shfl_xor(qs, m, 32);
        ks += __shfl_xor(ks, m, 32);
    }
    float qinv = rsqrtf(qs * (1.0f / 64.0f) + 1e-6f);
    float kinv = rsqrtf(ks * (1.0f / 64.0f) + 1e-6f);
    float qn0 = q0v * qinv * qw[d0], qn1 = q1v * qinv * qw[d1];
    float kn0 = k0v * kinv * kw[d0], kn1 = k1v * kinv * kw[d1];

    // RoPE on dims 0..31 (pairs (2j,2j+1)); lane covers exactly the rot dims.
    int j = lane >> 1;
    float c = cosT[t * 16 + j], s = sinT[t * 16 + j];
    float qo = __shfl_xor(qn0, 1, 32);
    float ko = __shfl_xor(kn0, 1, 32);
    float qr = (lane & 1) ? (qo * s + qn0 * c) : (qn0 * c - qo * s);
    float kr = (lane & 1) ? (ko * s + kn0 * c) : (kn0 * c - ko * s);

    size_t qkb = ((size_t)h * T_SEQ + t) * HD;
    qb[qkb + d0]   = (bf16)qr;  qb[qkb + d1]   = (bf16)qn1;
    kbuf[qkb + d0] = (bf16)kr;  kbuf[qkb + d1] = (bf16)kn1;
    size_t vb = (size_t)h * HD * T_SEQ + t;
    vt[vb + (size_t)d0 * T_SEQ] = (bf16)v0v;
    vt[vb + (size_t)d1 * T_SEQ] = (bf16)v1v;
}

// ---------------------------------------------------------------------------
// Flash attention with 1024-token sliding window.  One wave per 16-query
// tile; online softmax over 32-key blocks; WMMA for QK^T and PV; LDS used
// for the P C-layout -> A-layout transpose (per-wave buffer, DS in-order).
// attnb: [T][DM] bf16 (input to output projection GEMM)
// ---------------------------------------------------------------------------
__global__ __launch_bounds__(256) void attn_window_wmma(
    const bf16* __restrict__ qb, const bf16* __restrict__ kbuf,
    const bf16* __restrict__ vt, bf16* __restrict__ attnb) {
    __shared__ __bf16 sP[8][16 * 32];
    const int lane = threadIdx.x & 31;
    const int wave = threadIdx.x >> 5;
    const int h  = blockIdx.y;
    const int q0 = blockIdx.x * 128 + wave * 16;
    const int half = lane >> 4;
    const int ln = lane & 15;

    const bf16* qh = qb   + (size_t)h * T_SEQ * HD;
    const bf16* kh = kbuf + (size_t)h * T_SEQ * HD;
    const bf16* vh = vt   + (size_t)h * HD * T_SEQ;

    const bf16* qrow = qh + (size_t)(q0 + ln) * HD;
    v16bf qfl = load_frag(qrow, 0, lane);
    v16bf qfh = load_frag(qrow, 32, lane);

    v8f acc0 = {}, acc1 = {}, acc2 = {}, acc3 = {};
    float mrow[8], lrow[8];
#pragma unroll
    for (int i = 0; i < 8; ++i) { mrow[i] = NEG_BIG; lrow[i] = 0.0f; }

    int kb0 = (q0 > (WINDOW - 1)) ? ((q0 - (WINDOW - 1)) & ~31) : 0;
    for (int kk = kb0; kk < q0 + 16; kk += 32) {
        // ---- S = Q K^T for 32 keys (two 16-key tiles) ----
        v8f s0 = {}, s1 = {};
        {
            const bf16* kr0 = kh + (size_t)(kk + ln) * HD;
            const bf16* kr1 = kh + (size_t)(kk + 16 + ln) * HD;
            s0 = wmma_bf16(qfl, load_frag(kr0, 0, lane), s0);
            s0 = wmma_bf16(qfh, load_frag(kr0, 32, lane), s0);
            s1 = wmma_bf16(qfl, load_frag(kr1, 0, lane), s1);
            s1 = wmma_bf16(qfh, load_frag(kr1, 32, lane), s1);
        }
        // ---- scale + window mask ----
        float v0[8], v1[8], mx[8];
#pragma unroll
        for (int vi = 0; vi < 8; ++vi) {
            int qi = q0 + vi + 8 * half;
            int dd0 = qi - (kk + ln);
            int dd1 = qi - (kk + 16 + ln);
            v0[vi] = (dd0 >= 0 && dd0 < WINDOW) ? s0[vi] * SCALE_F : NEG_BIG;
            v1[vi] = (dd1 >= 0 && dd1 < WINDOW) ? s1[vi] * SCALE_F : NEG_BIG;
            mx[vi] = fmaxf(v0[vi], v1[vi]);
        }
#pragma unroll
        for (int m = 1; m <= 8; m <<= 1)
#pragma unroll
            for (int vi = 0; vi < 8; ++vi)
                mx[vi] = fmaxf(mx[vi], __shfl_xor(mx[vi], m, 32));
        // ---- online softmax update ----
        float p0[8], p1[8], rs[8], alpha[8];
#pragma unroll
        for (int vi = 0; vi < 8; ++vi) {
            float mn = fmaxf(mrow[vi], mx[vi]);
            alpha[vi] = __expf(mrow[vi] - mn);   // NEG-NEG -> 1, harmless (l=0,acc=0)
            p0[vi] = (v0[vi] <= NEG_BIG * 0.5f) ? 0.0f : __expf(v0[vi] - mn);
            p1[vi] = (v1[vi] <= NEG_BIG * 0.5f) ? 0.0f : __expf(v1[vi] - mn);
            mrow[vi] = mn;
            rs[vi] = p0[vi] + p1[vi];
        }
#pragma unroll
        for (int m = 1; m <= 8; m <<= 1)
#pragma unroll
            for (int vi = 0; vi < 8; ++vi)
                rs[vi] += __shfl_xor(rs[vi], m, 32);
#pragma unroll
        for (int vi = 0; vi < 8; ++vi) {
            lrow[vi] = lrow[vi] * alpha[vi] + rs[vi];
            acc0[vi] *= alpha[vi]; acc1[vi] *= alpha[vi];
            acc2[vi] *= alpha[vi]; acc3[vi] *= alpha[vi];
        }
        // ---- transpose P (C layout -> A layout) through per-wave LDS ----
#pragma unroll
        for (int vi = 0; vi < 8; ++vi) {
            int r = vi + 8 * half;
            sP[wave][r * 32 + ln]      = (bf16)p0[vi];
            sP[wave][r * 32 + 16 + ln] = (bf16)p1[vi];
        }
        v16bf pa;
        {
            const int base = half ? 8 : 0;
#pragma unroll
            for (int i = 0; i < 8; ++i) {
                pa[i]     = sP[wave][ln * 32 + base + i];
                pa[i + 8] = sP[wave][ln * 32 + 16 + base + i];
            }
        }
        // ---- acc += P x V  (V transposed: rows=dim, cols=key) ----
        acc0 = wmma_bf16(pa, load_frag(vh + (size_t)( 0 + ln) * T_SEQ, kk, lane), acc0);
        acc1 = wmma_bf16(pa, load_frag(vh + (size_t)(16 + ln) * T_SEQ, kk, lane), acc1);
        acc2 = wmma_bf16(pa, load_frag(vh + (size_t)(32 + ln) * T_SEQ, kk, lane), acc2);
        acc3 = wmma_bf16(pa, load_frag(vh + (size_t)(48 + ln) * T_SEQ, kk, lane), acc3);
    }
    // ---- normalize and emit [T][DM] bf16 ----
#pragma unroll
    for (int vi = 0; vi < 8; ++vi) {
        int t = q0 + vi + 8 * half;
        float inv = 1.0f / lrow[vi];
        size_t o = (size_t)t * DM + h * HD + ln;
        attnb[o + 0]  = (bf16)(acc0[vi] * inv);
        attnb[o + 16] = (bf16)(acc1[vi] * inv);
        attnb[o + 32] = (bf16)(acc2[vi] * inv);
        attnb[o + 48] = (bf16)(acc3[vi] * inv);
    }
}

// ---------------------------------------------------------------------------
extern "C" void kernel_launch(void* const* d_in, const int* in_sizes, int n_in,
                              void* d_out, int out_size, void* d_ws, size_t ws_size,
                              hipStream_t stream) {
    const float* x     = (const float*)d_in[0];
    const float* w_qkv = (const float*)d_in[3];
    const float* w_o   = (const float*)d_in[4];
    const float* qnw   = (const float*)d_in[5];
    const float* knw   = (const float*)d_in[6];
    const float* cosT  = (const float*)d_in[7];
    const float* sinT  = (const float*)d_in[8];

    char* ws = (char*)d_ws;
    size_t off = 0;
    auto alloc = [&](size_t bytes) {
        void* p = ws + off;
        off += (bytes + 255) & ~(size_t)255;
        return p;
    };
    bf16*  xb    = (bf16*) alloc((size_t)T_SEQ * DM * 2);          // 8 MB
    bf16*  wqkvb = (bf16*) alloc((size_t)3 * DM * DM * 2);         // 6 MB
    bf16*  wob   = (bf16*) alloc((size_t)DM * DM * 2);             // 2 MB
    float* qkv   = (float*)alloc((size_t)T_SEQ * 3 * DM * 4);      // 48 MB
    bf16*  qb    = (bf16*) alloc((size_t)NH * T_SEQ * HD * 2);     // 8 MB
    bf16*  kbuf  = (bf16*) alloc((size_t)NH * T_SEQ * HD * 2);     // 8 MB
    bf16*  vt    = (bf16*) alloc((size_t)NH * HD * T_SEQ * 2);     // 8 MB
    bf16*  attnb = (bf16*) alloc((size_t)T_SEQ * DM * 2);          // 8 MB

    int n1 = T_SEQ * DM, n2 = 3 * DM * DM, n3 = DM * DM;
    cvt_f32_bf16<<<(n1 + 255) / 256, 256, 0, stream>>>(x, xb, n1);
    cvt_f32_bf16<<<(n2 + 255) / 256, 256, 0, stream>>>(w_qkv, wqkvb, n2);
    cvt_f32_bf16<<<(n3 + 255) / 256, 256, 0, stream>>>(w_o, wob, n3);

    gemm_bf16_wmma<DM><<<dim3(3 * DM / 256, T_SEQ / 32), 256, 0, stream>>>(
        xb, wqkvb, qkv, T_SEQ, 3 * DM);

    norm_rope_pack<<<T_SEQ * NH / 8, 256, 0, stream>>>(
        qkv, qnw, knw, cosT, sinT, qb, kbuf, vt);

    attn_window_wmma<<<dim3(T_SEQ / 128, NH), 256, 0, stream>>>(
        qb, kbuf, vt, attnb);

    gemm_bf16_wmma<DM><<<dim3(DM / 256, T_SEQ / 32), 256, 0, stream>>>(
        attnb, wob, (float*)d_out, T_SEQ, DM);
}
